// DeepSpatialMultiTaskModel_57707180589300
// MI455X (gfx1250) — matrix operand order repaired and checked by reading, not compiled, not run on hardware
//
#include <hip/hip_runtime.h>
#include <hip/hip_bf16.h>
#include <cmath>

// ---------------------------------------------------------------------------
// DeepSpatialMultiTaskModel forward for MI455X (gfx1250).
// All dense math on V_WMMA_F32_16X16X32_BF16. GEMM staging uses the CDNA5
// async paths: per-lane GLOBAL_LOAD_ASYNC_TO_LDS_B128 for the A tile
// (ASYNCcnt) and the Tensor Data Mover (TENSOR_LOAD_TO_LDS, TENSORcnt) for
// the uniform B weight panel, with TDM row padding producing the
// bank-conflict-free LDS stride and TDM OOB zero-fill handling Nc tails.
// ---------------------------------------------------------------------------

#define N_NODES 8192
#define N_EDGES 131072
#define HID 256
#define EPSV 1e-5f

typedef unsigned short u16;
typedef unsigned int u32;
typedef __bf16 bf16raw;
typedef bf16raw v16bf __attribute__((ext_vector_type(16)));
typedef float v8f __attribute__((ext_vector_type(8)));
typedef u32 v4u __attribute__((ext_vector_type(4)));
typedef u32 v8u __attribute__((ext_vector_type(8)));

__device__ __forceinline__ u16 f2bf(float f) {
  union { float f; u32 u; } c; c.f = f;
  u32 r = c.u + 0x7FFFu + ((c.u >> 16) & 1u);   // round-to-nearest-even
  return (u16)(r >> 16);
}
__device__ __forceinline__ float bf2f(u16 h) {
  union { u32 u; float f; } c; c.u = ((u32)h) << 16;
  return c.f;
}
__device__ __forceinline__ float gelu_exact(float x) {
  return 0.5f * x * (1.0f + erff(x * 0.70710678118654752f));
}

// ---------------------------------------------------------------------------
// Generic WMMA GEMM:  out[M,Nc] = act( (A[M,K] * B[Nc,K]^T) * ga[n] + bb[n] )
// A row-major bf16, B row-major bf16 ([out,in] weight layout == native WMMA
// B-fragment order).  Modes: GELU, bf16 store, scatter-atomicAdd by row_map.
// ---------------------------------------------------------------------------
#define BM 128
#define BN 128
#define BK 32
#define LK 40                 // padded LDS row stride (elements) == 32 + 8

#define MODE_GELU    1
#define MODE_BF16    2
#define MODE_SCATTER 4

__global__ __launch_bounds__(256) void k_gemm_wmma(
    const u16* __restrict__ A, const u16* __restrict__ B,
    const float* __restrict__ ga, const float* __restrict__ bb,
    float* __restrict__ outF, u16* __restrict__ outH,
    const int* __restrict__ row_map,
    int M, int Nc, int K, int mode)
{
  __shared__ __align__(16) u16 As[2][BM * LK];
  __shared__ __align__(16) u16 Bs[2][BN * LK];

  const int bm = blockIdx.x * BM;
  const int bn = blockIdx.y * BN;
  const int tid  = threadIdx.x;
  const int wave = tid >> 5;
  const int lane = tid & 31;
  const int wm = (wave & 1) * 64;     // 2 wave rows  x 64
  const int wn = (wave >> 1) * 32;    // 4 wave cols  x 32
  const int lrow = lane & 15;
  const int hi   = lane >> 4;

  const unsigned ldsA = (unsigned)(size_t)(&As[0][0]);   // LDS byte offset
  const unsigned ldsB = (unsigned)(size_t)(&Bs[0][0]);

  // Pre-zero A rows past M (kept for generality; all launches have M%128==0).
  if (bm + BM > M) {
    for (int c = tid; c < 512; c += 256) {
      int row = c >> 2, col = (c & 3) * 8;
      if (bm + row >= M) {
        uint4 z = make_uint4(0u, 0u, 0u, 0u);
        *(uint4*)(&As[0][row * LK + col]) = z;
        *(uint4*)(&As[1][row * LK + col]) = z;
      }
    }
  }

  auto load_stage = [&](int s, int k0) {
    // ---- A tile: per-lane async global->LDS (ASYNCcnt), 16B chunks ----
#pragma unroll
    for (int i = 0; i < 2; ++i) {
      int c = tid + i * 256;
      int row = c >> 2, col = (c & 3) * 8;
      int gr = bm + row;
      if (gr < M) {
        unsigned loff = ldsA + (unsigned)((s * BM * LK + row * LK + col) * 2);
        const u16* p = A + (size_t)gr * K + (k0 + col);
        asm volatile("global_load_async_to_lds_b128 %0, %1, off"
                     :: "v"(loff), "v"(p) : "memory");
        if (k0 + BK < K) __builtin_prefetch(p + BK, 0, 3);  // global_prefetch_b8
      }
    }
    // ---- B tile: Tensor Data Mover, one descriptor per stage (wave 0) ----
    // 2-D tile: tile_dim0 = 32 elems (64B rows), tile_dim1 = 128 rows,
    // tensor_dim0_stride = K.  pad_interval=16 DW + pad_amount=4 DW appends
    // 16B per row -> LDS row stride of 40 elements.  tensor_dim1 = Nc-bn so
    // OOB weight rows are zero-filled by the TDM.
    if (wave == 0) {
      unsigned long long gaddr =
          (unsigned long long)(size_t)(B + (size_t)bn * K + k0);
      unsigned lds_b = ldsB + (unsigned)(s * BN * LK * 2);
      u32 d0 = (u32)(K - k0);          // remaining elements in dim0
      u32 d1 = (u32)(Nc - bn);         // remaining rows in dim1
      v4u g0;
      g0[0] = 1u;                                  // count=1, user mode
      g0[1] = lds_b;                               // lds_addr
      g0[2] = (u32)gaddr;                          // global_addr[31:0]
      g0[3] = (u32)(gaddr >> 32) | (2u << 30);     // global_addr hi | type=2
      v8u g1;
      g1[0] = (1u << 16) | (1u << 20) | (3u << 22) | (3u << 25);
              // data_size=2B, pad_enable, pad_interval=16DW, pad_amount=4DW
      g1[1] = (d0 & 0xFFFFu) << 16;                // tensor_dim0[15:0]
      g1[2] = (d0 >> 16) | ((d1 & 0xFFFFu) << 16); // dim0 hi | tensor_dim1 lo
      g1[3] = (d1 >> 16) | (32u << 16);            // dim1 hi | tile_dim0=32
      g1[4] = 128u;                                // tile_dim1=128
      g1[5] = (u32)K;                              // tensor_dim0_stride lo
      g1[6] = 0u;
      g1[7] = 0u;
      asm volatile("tensor_load_to_lds %0, %1" :: "s"(g0), "s"(g1) : "memory");
    }
  };

  auto drain_stage = [&]() {
    asm volatile("s_wait_asynccnt 0x0" ::: "memory");
    if (wave == 0) __builtin_amdgcn_s_wait_tensorcnt(0);
    __syncthreads();
  };

  v8f acc[4][2];
#pragma unroll
  for (int mt = 0; mt < 4; ++mt)
#pragma unroll
    for (int nt = 0; nt < 2; ++nt)
#pragma unroll
      for (int r = 0; r < 8; ++r) acc[mt][nt][r] = 0.0f;

  load_stage(0, 0);
  drain_stage();

  const int KT = K / BK;
  for (int kt = 0; kt < KT; ++kt) {
    const int s = kt & 1;
    if (kt + 1 < KT) load_stage(s ^ 1, (kt + 1) * BK);

    union Frag { uint4 u[2]; v16bf v; };
    Frag af[4], bfr[2];
    // A fragment (16x32 bf16): lanes 0-15 -> K chunks [0..7],[16..23];
    // lanes 16-31 -> [8..15],[24..31]  (ISA 7.12.2)
#pragma unroll
    for (int mt = 0; mt < 4; ++mt) {
      const u16* p = &As[s][(wm + mt * 16 + lrow) * LK + hi * 8];
      af[mt].u[0] = *(const uint4*)p;
      af[mt].u[1] = *(const uint4*)(p + 16);
    }
    // B fragment (32x16 bf16): same K chunks as A for column lrow
#pragma unroll
    for (int nt = 0; nt < 2; ++nt) {
      const u16* p = &Bs[s][(wn + nt * 16 + lrow) * LK + hi * 8];
      bfr[nt].u[0] = *(const uint4*)p;
      bfr[nt].u[1] = *(const uint4*)(p + 16);
    }
#pragma unroll
    for (int mt = 0; mt < 4; ++mt)
#pragma unroll
      for (int nt = 0; nt < 2; ++nt)
        acc[mt][nt] = __builtin_amdgcn_wmma_f32_16x16x32_bf16(
            false, af[mt].v, false, bfr[nt].v, (short)0, acc[mt][nt],
            false, false);
    drain_stage();
  }

  // Epilogue: C/D layout -> row = base + hi*8 + r, col = base + lrow
#pragma unroll
  for (int nt = 0; nt < 2; ++nt) {
    int gc = bn + wn + nt * 16 + lrow;
    if (gc >= Nc) continue;
    float gsc = ga ? ga[gc] : 1.0f;
    float bof = bb ? bb[gc] : 0.0f;
#pragma unroll
    for (int mt = 0; mt < 4; ++mt) {
#pragma unroll
      for (int r = 0; r < 8; ++r) {
        int gr = bm + wm + mt * 16 + hi * 8 + r;
        if (gr >= M) continue;
        float v = acc[mt][nt][r] * gsc + bof;
        if (mode & MODE_GELU) v = gelu_exact(v);
        if (mode & MODE_SCATTER) {
          atomicAdd(outF + (size_t)row_map[gr] * Nc + gc, v);
        } else if (mode & MODE_BF16) {
          outH[(size_t)gr * Nc + gc] = f2bf(v);
        } else {
          outF[(size_t)gr * Nc + gc] = v;
        }
      }
    }
  }
}

// ---------------------------------------------------------------------------
// im2col: build [M, Kp] bf16 A-matrix. srcF = f32 NCHW (first conv) or
// srcH = bf16 NHWC (previous conv output, which is GEMM row-major [n*oh*ow,C]).
// k = ci*KH*KW + ky*KW + kx  (matches OIHW weight flattening).
// ---------------------------------------------------------------------------
__global__ void k_im2col(u16* __restrict__ dst, const u16* __restrict__ srcH,
                         const float* __restrict__ srcF,
                         int Nn, int C, int H, int W, int KH, int KW,
                         int stride, int pad, int OH, int OW, int K, int Kp,
                         long long total)
{
  long long idx = (long long)blockIdx.x * blockDim.x + threadIdx.x;
  if (idx >= total) return;
  int k = (int)(idx % Kp);
  long long m = idx / Kp;
  u16 v = 0;
  if (k < K) {
    int kx = k % KW; int t = k / KW; int ky = t % KH; int ci = t / KH;
    int ox = (int)(m % OW); long long t2 = m / OW;
    int oy = (int)(t2 % OH); int n = (int)(t2 / OH);
    int iy = oy * stride - pad + ky;
    int ix = ox * stride - pad + kx;
    if (iy >= 0 && iy < H && ix >= 0 && ix < W) {
      if (srcF) v = f2bf(srcF[(((size_t)n * C + ci) * H + iy) * W + ix]);
      else      v = srcH[(((size_t)n * H + iy) * W + ix) * C + ci];
    }
  }
  dst[idx] = v;
}

// weight pack: f32 [R,K] -> bf16 [R,Kp] zero-padded
__global__ void k_pack_w(u16* __restrict__ dst, const float* __restrict__ src,
                         int R, int K, int Kp)
{
  int idx = blockIdx.x * blockDim.x + threadIdx.x;
  if (idx >= R * Kp) return;
  int k = idx % Kp, r = idx / Kp;
  dst[idx] = (k < K) ? f2bf(src[(size_t)r * K + k]) : (u16)0;
}

// fold eval-mode BN into scale/bias:  y = conv*ga + bb
__global__ void k_prep_bn(float* ga, float* bb, const float* b,
                          const float* g, const float* beta, int C)
{
  int c = blockIdx.x * blockDim.x + threadIdx.x;
  if (c >= C) return;
  float s = rsqrtf(1.0f + EPSV);
  float gs = g[c] * s;
  ga[c] = gs;
  bb[c] = b[c] * gs + beta[c];
}

__global__ void k_prep_headb(float* hb, const float* cb, const float* mb,
                             const float* pb)
{
  int c = threadIdx.x;
  if (c >= 64) return;
  float v = 0.0f;
  if (c < 8)       v = cb[c];
  else if (c < 24) v = mb[c - 8];
  else if (c < 56) v = pb[c - 24];
  hb[c] = v;
}

// adaptive avg-pool over the 2x2 spatial grid of conv4 output
__global__ void k_pool4(u16* __restrict__ h, const u16* __restrict__ c4,
                        int total)
{
  int idx = blockIdx.x * blockDim.x + threadIdx.x;
  if (idx >= total) return;
  int c = idx & 255; int n = idx >> 8;
  float s = 0.0f;
#pragma unroll
  for (int p = 0; p < 4; ++p) s += bf2f(c4[((size_t)n * 4 + p) * 256 + c]);
  h[idx] = f2bf(s * 0.25f);
}

// projector input: [img_feat | coords/1e4 | zero pad] as bf16 [N,288]
__global__ void k_build_xin(u16* __restrict__ xin,
                            const float* __restrict__ img_feat,
                            const float* __restrict__ coords, int total)
{
  int idx = blockIdx.x * blockDim.x + threadIdx.x;
  if (idx >= total) return;
  int c = idx % 288; int n = idx / 288;
  u16 v = 0;
  if (c < 256)      v = f2bf(img_feat[(size_t)n * 256 + c]);
  else if (c < 258) v = f2bf(coords[(size_t)n * 2 + (c - 256)] * 1e-4f);
  xin[idx] = v;
}

__global__ void k_deg(float* __restrict__ deg, const int* __restrict__ dst,
                      int E)
{
  int e = blockIdx.x * blockDim.x + threadIdx.x;
  if (e < E) atomicAdd(deg + dst[e], 1.0f);
}

__global__ void k_invdeg(float* __restrict__ inv, const float* __restrict__ deg,
                         int n)
{
  int i = blockIdx.x * blockDim.x + threadIdx.x;
  if (i < n) inv[i] = 1.0f / fmaxf(deg[i], 1.0f);
}

// edge gather: dst[e,:] = x_bf16[src[e],:]  (16B chunks)
__global__ void k_gather(u16* __restrict__ dst, const u16* __restrict__ x,
                         const int* __restrict__ src, int total)
{
  int idx = blockIdx.x * blockDim.x + threadIdx.x;
  if (idx >= total) return;
  int ch = idx & 31; int e = idx >> 5;
  ((uint4*)dst)[idx] = ((const uint4*)(x + (size_t)src[e] * HID))[ch];
}

// -------- row reductions (256 threads, wave32) --------
__device__ __forceinline__ float block_sum256(float v, float* red) {
#pragma unroll
  for (int o = 16; o > 0; o >>= 1) v += __shfl_xor(v, o, 32);
  int w = threadIdx.x >> 5, l = threadIdx.x & 31;
  if (l == 0) red[w] = v;
  __syncthreads();
  float t = 0.0f;
#pragma unroll
  for (int i = 0; i < 8; ++i) t += red[i];
  __syncthreads();
  return t;
}

// LayerNorm (+ optional GELU), writes f32 + bf16 copies
__global__ void __launch_bounds__(256) k_ln_act(
    const float* __restrict__ in, const float* __restrict__ g,
    const float* __restrict__ b, float* __restrict__ xf, u16* __restrict__ xh,
    int gelu_flag)
{
  __shared__ float red[8];
  int row = blockIdx.x, tid = threadIdx.x;
  size_t base = (size_t)row * HID;
  float v  = in[base + tid];
  float mu = block_sum256(v, red) * (1.0f / HID);
  float d  = v - mu;
  float var = block_sum256(d * d, red) * (1.0f / HID);
  float y = d * rsqrtf(var + EPSV) * g[tid] + b[tid];
  if (gelu_flag) y = gelu_exact(y);
  xf[base + tid] = y;
  xh[base + tid] = f2bf(y);
}

// graph block tail: out = LN( x + gelu(sp + agg*inv_deg) )
__global__ void __launch_bounds__(256) k_combine_ln(
    const float* __restrict__ sp, const float* __restrict__ agg,
    const float* __restrict__ invdeg, const float* __restrict__ xin,
    const float* __restrict__ g, const float* __restrict__ b,
    float* __restrict__ xf_out, u16* __restrict__ xh_out)
{
  __shared__ float red[8];
  int row = blockIdx.x, tid = threadIdx.x;
  size_t base = (size_t)row * HID;
  float a = sp[base + tid] + agg[base + tid] * invdeg[row];
  float t = xin[base + tid] + gelu_exact(a);
  float mu = block_sum256(t, red) * (1.0f / HID);
  float d  = t - mu;
  float var = block_sum256(d * d, red) * (1.0f / HID);
  float y = d * rsqrtf(var + EPSV) * g[tid] + b[tid];
  xf_out[base + tid] = y;
  xh_out[base + tid] = f2bf(y);
}

// split fused head GEMM [N,64] -> comp[N,8] | cmpo[N,16] | prog[N,32]
__global__ void k_split_heads(const float* __restrict__ ho,
                              float* __restrict__ out, int total)
{
  int idx = blockIdx.x * blockDim.x + threadIdx.x;
  if (idx >= total) return;
  int c = idx % 56; int n = idx / 56;
  float v = ho[(size_t)n * 64 + c];
  if (c < 8)
    out[(size_t)N_NODES * 256 + (size_t)n * 8 + c] = v;
  else if (c < 24)
    out[(size_t)N_NODES * 264 + (size_t)n * 16 + (c - 8)] = v;
  else
    out[(size_t)N_NODES * 280 + (size_t)n * 32 + (c - 24)] = v;
}

// ---------------------------------------------------------------------------
extern "C" void kernel_launch(void* const* d_in, const int* in_sizes, int n_in,
                              void* d_out, int out_size, void* d_ws,
                              size_t ws_size, hipStream_t stream)
{
  (void)in_sizes; (void)n_in; (void)out_size; (void)ws_size;

  const float* patches    = (const float*)d_in[0];
  const float* coords     = (const float*)d_in[1];
  const int*   edge_index = (const int*)d_in[2];
  const float* c1_w = (const float*)d_in[3];  const float* c1_b = (const float*)d_in[4];
  const float* bn1g = (const float*)d_in[5];  const float* bn1b = (const float*)d_in[6];
  const float* c2_w = (const float*)d_in[7];  const float* c2_b = (const float*)d_in[8];
  const float* bn2g = (const float*)d_in[9];  const float* bn2b = (const float*)d_in[10];
  const float* c3_w = (const float*)d_in[11]; const float* c3_b = (const float*)d_in[12];
  const float* bn3g = (const float*)d_in[13]; const float* bn3b = (const float*)d_in[14];
  const float* c4_w = (const float*)d_in[15]; const float* c4_b = (const float*)d_in[16];
  const float* bn4g = (const float*)d_in[17]; const float* bn4b = (const float*)d_in[18];
  const float* enc_w = (const float*)d_in[19]; const float* enc_b = (const float*)d_in[20];
  const float* pj_w  = (const float*)d_in[21]; const float* pj_b  = (const float*)d_in[22];
  const float* plng  = (const float*)d_in[23]; const float* plnb  = (const float*)d_in[24];
  const float* msg_w = (const float*)d_in[25]; const float* msg_b = (const float*)d_in[26];
  const float* sp_w  = (const float*)d_in[27]; const float* sp_b  = (const float*)d_in[28];
  const float* lng   = (const float*)d_in[29]; const float* lnb   = (const float*)d_in[30];
  const float* comp_w = (const float*)d_in[31]; const float* comp_b = (const float*)d_in[32];
  const float* cmpo_w = (const float*)d_in[33]; const float* cmpo_b = (const float*)d_in[34];
  const float* prog_w = (const float*)d_in[35]; const float* prog_b = (const float*)d_in[36];

  const int* e_src = edge_index;
  const int* e_dst = edge_index + N_EDGES;
  float* out = (float*)d_out;

  // ---- workspace carve-out ----
  char* ws = (char*)d_ws;
  size_t off = 0;
  auto alloc = [&](size_t bytes) -> char* {
    char* p = ws + off;
    off = (off + bytes + 255) & ~(size_t)255;
    return p;
  };
  u16* w1   = (u16*)alloc((size_t)32 * 96 * 2);
  u16* w2   = (u16*)alloc((size_t)64 * 288 * 2);
  u16* w3   = (u16*)alloc((size_t)128 * 576 * 2);
  u16* w4   = (u16*)alloc((size_t)256 * 1152 * 2);
  u16* wenc = (u16*)alloc((size_t)256 * 256 * 2);
  u16* wpj  = (u16*)alloc((size_t)256 * 288 * 2);
  u16* wmsg = (u16*)alloc((size_t)4 * 256 * 256 * 2);
  u16* wsp  = (u16*)alloc((size_t)4 * 256 * 256 * 2);
  u16* whd  = (u16*)alloc((size_t)64 * 256 * 2);
  float* ga1 = (float*)alloc(32 * 4);  float* bb1 = (float*)alloc(32 * 4);
  float* ga2 = (float*)alloc(64 * 4);  float* bb2 = (float*)alloc(64 * 4);
  float* ga3 = (float*)alloc(128 * 4); float* bb3 = (float*)alloc(128 * 4);
  float* ga4 = (float*)alloc(256 * 4); float* bb4 = (float*)alloc(256 * 4);
  float* hdb = (float*)alloc(64 * 4);
  float* deg = (float*)alloc((size_t)N_NODES * 4);
  float* inv = (float*)alloc((size_t)N_NODES * 4);
  u16* IM   = (u16*)alloc((size_t)2097152 * 96 * 2);     // largest im2col / gather
  u16* OUTA = (u16*)alloc((size_t)2097152 * 32 * 2);
  u16* OUTB = (u16*)alloc((size_t)524288 * 64 * 2);
  u16* hbuf = (u16*)alloc((size_t)N_NODES * 256 * 2);
  float* imgf = (float*)alloc((size_t)N_NODES * 256 * 4);
  u16* xin    = (u16*)alloc((size_t)N_NODES * 288 * 2);
  float* lin  = (float*)alloc((size_t)N_NODES * 256 * 4);
  float* xf   = (float*)alloc((size_t)N_NODES * 256 * 4);
  u16* xh     = (u16*)alloc((size_t)N_NODES * 256 * 2);
  float* agg  = (float*)alloc((size_t)N_NODES * 256 * 4);
  float* hout = (float*)alloc((size_t)N_NODES * 64 * 4);

  auto gemm = [&](const u16* A, const u16* B, const float* ga, const float* bb,
                  float* oF, u16* oH, const int* rmap, int M, int Nc, int K,
                  int mode) {
    dim3 g((M + BM - 1) / BM, (Nc + BN - 1) / BN);
    k_gemm_wmma<<<g, dim3(256), 0, stream>>>(A, B, ga, bb, oF, oH, rmap, M, Nc,
                                             K, mode);
  };
  auto grid1 = [](long long total) {
    return dim3((unsigned)((total + 255) / 256));
  };

  // ---- weight packing + BN folding ----
  k_pack_w<<<grid1(32 * 96), 256, 0, stream>>>(w1, c1_w, 32, 75, 96);
  k_pack_w<<<grid1(64 * 288), 256, 0, stream>>>(w2, c2_w, 64, 288, 288);
  k_pack_w<<<grid1(128 * 576), 256, 0, stream>>>(w3, c3_w, 128, 576, 576);
  k_pack_w<<<grid1(256 * 1152), 256, 0, stream>>>(w4, c4_w, 256, 1152, 1152);
  k_pack_w<<<grid1(256 * 256), 256, 0, stream>>>(wenc, enc_w, 256, 256, 256);
  k_pack_w<<<grid1(256 * 288), 256, 0, stream>>>(wpj, pj_w, 256, 258, 288);
  for (int l = 0; l < 4; ++l) {
    k_pack_w<<<grid1(256 * 256), 256, 0, stream>>>(wmsg + (size_t)l * 65536,
                                                   msg_w + (size_t)l * 65536,
                                                   256, 256, 256);
    k_pack_w<<<grid1(256 * 256), 256, 0, stream>>>(wsp + (size_t)l * 65536,
                                                   sp_w + (size_t)l * 65536,
                                                   256, 256, 256);
  }
  hipMemsetAsync(whd, 0, (size_t)64 * 256 * 2, stream);
  k_pack_w<<<grid1(8 * 256), 256, 0, stream>>>(whd, comp_w, 8, 256, 256);
  k_pack_w<<<grid1(16 * 256), 256, 0, stream>>>(whd + (size_t)8 * 256, cmpo_w, 16, 256, 256);
  k_pack_w<<<grid1(32 * 256), 256, 0, stream>>>(whd + (size_t)24 * 256, prog_w, 32, 256, 256);
  k_prep_bn<<<1, 256, 0, stream>>>(ga1, bb1, c1_b, bn1g, bn1b, 32);
  k_prep_bn<<<1, 256, 0, stream>>>(ga2, bb2, c2_b, bn2g, bn2b, 64);
  k_prep_bn<<<1, 256, 0, stream>>>(ga3, bb3, c3_b, bn3g, bn3b, 128);
  k_prep_bn<<<1, 256, 0, stream>>>(ga4, bb4, c4_b, bn4g, bn4b, 256);
  k_prep_headb<<<1, 64, 0, stream>>>(hdb, comp_b, cmpo_b, prog_b);

  // ---- degree / inverse degree ----
  hipMemsetAsync(deg, 0, (size_t)N_NODES * 4, stream);
  k_deg<<<grid1(N_EDGES), 256, 0, stream>>>(deg, e_dst, N_EDGES);
  k_invdeg<<<grid1(N_NODES), 256, 0, stream>>>(inv, deg, N_NODES);

  // ---- CNN encoder: im2col + WMMA GEMM (+folded BN + GELU, bf16 out) ----
  long long t1 = (long long)2097152 * 96;
  k_im2col<<<grid1(t1), 256, 0, stream>>>(IM, nullptr, patches, 8192, 3, 32, 32,
                                          5, 5, 2, 2, 16, 16, 75, 96, t1);
  gemm(IM, w1, ga1, bb1, nullptr, OUTA, nullptr, 2097152, 32, 96,
       MODE_GELU | MODE_BF16);

  long long t2 = (long long)524288 * 288;
  k_im2col<<<grid1(t2), 256, 0, stream>>>(IM, OUTA, nullptr, 8192, 32, 16, 16,
                                          3, 3, 2, 1, 8, 8, 288, 288, t2);
  gemm(IM, w2, ga2, bb2, nullptr, OUTB, nullptr, 524288, 64, 288,
       MODE_GELU | MODE_BF16);

  long long t3 = (long long)131072 * 576;
  k_im2col<<<grid1(t3), 256, 0, stream>>>(IM, OUTB, nullptr, 8192, 64, 8, 8,
                                          3, 3, 2, 1, 4, 4, 576, 576, t3);
  gemm(IM, w3, ga3, bb3, nullptr, OUTA, nullptr, 131072, 128, 576,
       MODE_GELU | MODE_BF16);

  long long t4 = (long long)32768 * 1152;
  k_im2col<<<grid1(t4), 256, 0, stream>>>(IM, OUTA, nullptr, 8192, 128, 4, 4,
                                          3, 3, 2, 1, 2, 2, 1152, 1152, t4);
  gemm(IM, w4, ga4, bb4, nullptr, OUTB, nullptr, 32768, 256, 1152,
       MODE_GELU | MODE_BF16);

  k_pool4<<<grid1((long long)N_NODES * 256), 256, 0, stream>>>(
      hbuf, OUTB, N_NODES * 256);
  gemm(hbuf, wenc, nullptr, enc_b, imgf, nullptr, nullptr, N_NODES, 256, 256, 0);

  // ---- projector ----
  k_build_xin<<<grid1((long long)N_NODES * 288), 256, 0, stream>>>(
      xin, imgf, coords, N_NODES * 288);
  gemm(xin, wpj, nullptr, pj_b, lin, nullptr, nullptr, N_NODES, 256, 288, 0);
  k_ln_act<<<N_NODES, 256, 0, stream>>>(lin, plng, plnb, xf, xh, 1);

  // ---- graph blocks ----
  for (int l = 0; l < 4; ++l) {
    hipMemsetAsync(agg, 0, (size_t)N_NODES * 256 * 4, stream);
    k_gather<<<grid1((long long)N_EDGES * 32), 256, 0, stream>>>(
        IM, xh, e_src, N_EDGES * 32);
    // message GEMM, scatter-add (bias included) directly into agg by dst idx
    gemm(IM, wmsg + (size_t)l * 65536, nullptr, msg_b + (size_t)l * 256, agg,
         nullptr, e_dst, N_EDGES, 256, 256, MODE_SCATTER);
    // self-path GEMM
    gemm(xh, wsp + (size_t)l * 65536, nullptr, sp_b + (size_t)l * 256, lin,
         nullptr, nullptr, N_NODES, 256, 256, 0);
    k_combine_ln<<<N_NODES, 256, 0, stream>>>(
        lin, agg, inv, xf, lng + (size_t)l * 256, lnb + (size_t)l * 256,
        (l == 3) ? out : xf, xh);
  }

  // ---- heads (fused [64,256] weight, then split) ----
  gemm(xh, whd, nullptr, hdb, hout, nullptr, nullptr, N_NODES, 64, 256, 0);
  k_split_heads<<<grid1((long long)N_NODES * 56), 256, 0, stream>>>(
      hout, out, N_NODES * 56);
}
